// ATTCNN_84988812853677
// MI455X (gfx1250) — compile-verified, matrix operand order checked by reading, not run on hardware
//
#include <hip/hip_runtime.h>
#include <hip/hip_bf16.h>
#include <math.h>

#define B_   128
#define L_   128
#define E_   300
#define P_   50
#define FN_  256
#define FD_  1000
#define FDP_ 1024
#define H2_  100
#define LAB_ 19
#define OD_  1900   // 6*E + H2

typedef __attribute__((ext_vector_type(16))) _Float16 v16h;
typedef __attribute__((ext_vector_type(2)))  _Float16 v2h;
typedef __attribute__((ext_vector_type(8)))  float    v8f;

union FragH { v16h v; v2h h2[8]; };

#if __has_builtin(__builtin_amdgcn_global_load_async_to_lds_b128) && \
    __has_builtin(__builtin_amdgcn_s_wait_asynccnt)
#define USE_ASYNC_LDS 1
typedef int v4i_ __attribute__((vector_size(16)));
// 16B/lane Global -> LDS, ASYNCcnt-tracked; OFF applies to both addresses.
// Frontend signature (from probe): (v4i* gsrc, v4i* ldst, imm offset, imm cpol)
#define ASYNC_CP16(gp, lp, OFF)                                               \
    __builtin_amdgcn_global_load_async_to_lds_b128(                           \
        (v4i_*)(gp), (v4i_*)(lp), (OFF), 0)
#endif

// -------------------------------------------------------------------------
// 1) Build f16 embed matrix: (B*L + 8 pad rows) x 1024
//    row = b*L + t ; cols [0,900) = word window (zeroed at t==0),
//    [900,950) = pos1, [950,1000) = pos2, [1000,1024) = zero pad.
// -------------------------------------------------------------------------
__global__ void build_embed_kernel(const int* __restrict__ inputs,
                                   const int* __restrict__ p1,
                                   const int* __restrict__ p2,
                                   const float* __restrict__ emb,
                                   const float* __restrict__ pos1,
                                   const float* __restrict__ pos2,
                                   _Float16* __restrict__ embedH) {
    const int row = blockIdx.x;
    _Float16* d = embedH + (size_t)row * FDP_;
    if (row >= B_ * L_) {                       // zero pad rows (GEMM overreach)
        for (int c = threadIdx.x; c < FDP_; c += blockDim.x) d[c] = (_Float16)0.f;
        return;
    }
    const int b = row / L_, t = row % L_;
    const int q1 = p1[b * L_ + t], q2 = p2[b * L_ + t];
    for (int c = threadIdx.x; c < FDP_; c += blockDim.x) {
        float v = 0.f;
        if (c < 3 * E_) {
            if (t != 0) {                        // WF[:,0] = 0 in reference
                const int win = c / E_;
                const int e   = c - win * E_;
                const int pos = t + win;         // index into padded inp (>=1 here)
                const int tok = (pos <= L_) ? inputs[b * L_ + pos - 1] : 0;
                v = emb[(size_t)tok * E_ + e];
            }
        } else if (c < 3 * E_ + P_) {
            v = pos1[(size_t)q1 * P_ + (c - 3 * E_)];
        } else if (c < FD_) {
            v = pos2[(size_t)q2 * P_ + (c - 3 * E_ - P_)];
        }
        d[c] = (_Float16)v;
    }
}

// -------------------------------------------------------------------------
// 2) Repack conv weights f32 (FN,1,k,FD) -> f16 (FN, k*1024), zero-padded
// -------------------------------------------------------------------------
__global__ void repack_w_kernel(const float* __restrict__ w,
                                _Float16* __restrict__ dst, int kTaps) {
    const int rowId = blockIdx.x;               // 0 .. FN*k-1
    const float*  s = w   + (size_t)rowId * FD_;
    _Float16*     d = dst + (size_t)rowId * FDP_;
    for (int c = threadIdx.x; c < FDP_; c += blockDim.x)
        d[c] = (c < FD_) ? (_Float16)s[c] : (_Float16)0.f;
}

// -------------------------------------------------------------------------
// 3) Implicit-GEMM conv + tanh + max-over-position, one batch row per block.
//    Block tile: M=128 (positions t) x N=64 (filters); 8 waves (4 M x 2 N),
//    each wave 32x32 via 2x2 v_wmma_f32_16x16x32_f16 fragments.
//    K advances 64 halfs per stage; staging uses gfx1250 async Global->LDS
//    (ASYNCcnt) so no VGPR round-trip and no per-load loadcnt stalls.
// -------------------------------------------------------------------------
#define KSTEP_ 64   // halfs staged per iteration
#define LDA_   72   // halfs per LDS A row (144B stride, 16B-aligned sublanes)
#define LDB_   72

__global__ __launch_bounds__(256)
void conv_gemm_kernel(const _Float16* __restrict__ embedH,
                      const _Float16* __restrict__ wH,
                      const float* __restrict__ convB,
                      float* __restrict__ sf,
                      int kTaps, int sfOff) {
    __shared__ _Float16 As[128 * LDA_];
    __shared__ _Float16 Bs[64 * LDB_];
    __shared__ float    red[4 * 64];

    const int tid   = threadIdx.x;
    const int lane  = tid & 31;
    const int wave  = tid >> 5;
    const int wm    = wave & 3;        // M quadrant (rows 32*wm..)
    const int wn    = wave >> 2;       // N half    (cols 32*wn..)
    const int lrow  = lane & 15;
    const int lhi   = lane >> 4;

    const int b      = blockIdx.y;
    const int nTile0 = blockIdx.x * 64;
    const int KT     = kTaps * FDP_;              // halfs per weight row
    const int nsteps = kTaps * (FDP_ / KSTEP_);   // iterations of KSTEP_ halfs
    const int tmax   = L_ - kTaps;                // last valid conv position

    const _Float16* aBase = embedH + (size_t)b * L_ * FDP_;
    const _Float16* bBase = wH + (size_t)nTile0 * KT;

    // staging maps (uniform across loop)
    const int arow = tid >> 1, aoff = (tid & 1) * 32;   // 2 thr/row, 64B each
    const int brow = tid >> 2, boff = (tid & 3) * 16;   // 4 thr/row, 32B each
    _Float16* const aLds = &As[arow * LDA_ + aoff];
    _Float16* const bLds = &Bs[brow * LDB_ + boff];

    v8f c00 = {}, c01 = {}, c10 = {}, c11 = {};

    for (int s = 0; s < nsteps; ++s) {
        __syncthreads();
        const _Float16* ga = aBase + (size_t)arow * FDP_ + s * KSTEP_ + aoff;
        const _Float16* gb = bBase + (size_t)brow * KT   + s * KSTEP_ + boff;
#ifdef USE_ASYNC_LDS
        // ---- async stage: A 128x64 halfs (4x16B/lane), B 64x64 halfs (2x16B)
        ASYNC_CP16(ga, aLds, 0);
        ASYNC_CP16(ga, aLds, 16);
        ASYNC_CP16(ga, aLds, 32);
        ASYNC_CP16(ga, aLds, 48);
        ASYNC_CP16(gb, bLds, 0);
        ASYNC_CP16(gb, bLds, 16);
        if (s + 1 < nsteps) {   // gfx1250 global_prefetch_b8 for next K tile
            __builtin_prefetch(ga + KSTEP_, 0, 1);
            __builtin_prefetch(gb + KSTEP_, 0, 1);
        }
        __builtin_amdgcn_s_wait_asynccnt(0);
#else
        {
            const uint4* src = reinterpret_cast<const uint4*>(ga);
            uint4 d0 = src[0], d1 = src[1], d2 = src[2], d3 = src[3];
            uint4* dst = reinterpret_cast<uint4*>(aLds);
            dst[0] = d0; dst[1] = d1; dst[2] = d2; dst[3] = d3;
        }
        {
            const uint4* src = reinterpret_cast<const uint4*>(gb);
            uint4 d0 = src[0], d1 = src[1];
            uint4* dst = reinterpret_cast<uint4*>(bLds);
            dst[0] = d0; dst[1] = d1;
        }
        if (s + 1 < nsteps) {
            __builtin_prefetch(ga + KSTEP_, 0, 1);
            __builtin_prefetch(gb + KSTEP_, 0, 1);
        }
#endif
        __syncthreads();

        // ---- two 32-deep WMMA K-chunks out of the 64-half stage
#pragma unroll
        for (int q = 0; q < 2; ++q) {
            // CDNA5 16-bit A/B lane layout:
            // half-pair v holds K = 16*(v>>2) + 8*(lane>>4) + 2*(v&3)
            FragH a0, a1, b0, b1;
#pragma unroll
            for (int v = 0; v < 8; ++v) {
                const int kk = q * 32 + ((v >> 2) << 4) + (lhi << 3) + ((v & 3) << 1);
                a0.h2[v] = *(const v2h*)&As[(wm * 32 + lrow)      * LDA_ + kk];
                a1.h2[v] = *(const v2h*)&As[(wm * 32 + 16 + lrow) * LDA_ + kk];
                b0.h2[v] = *(const v2h*)&Bs[(wn * 32 + lrow)      * LDB_ + kk];
                b1.h2[v] = *(const v2h*)&Bs[(wn * 32 + 16 + lrow) * LDB_ + kk];
            }
            c00 = __builtin_amdgcn_wmma_f32_16x16x32_f16(false, a0.v, false, b0.v,
                                                         (short)0, c00, false, false);
            c01 = __builtin_amdgcn_wmma_f32_16x16x32_f16(false, a0.v, false, b1.v,
                                                         (short)0, c01, false, false);
            c10 = __builtin_amdgcn_wmma_f32_16x16x32_f16(false, a1.v, false, b0.v,
                                                         (short)0, c10, false, false);
            c11 = __builtin_amdgcn_wmma_f32_16x16x32_f16(false, a1.v, false, b1.v,
                                                         (short)0, c11, false, false);
        }
    }

    // ---- epilogue: z = tanh(acc + bias[n]); max over valid positions t
    // C/D layout: reg r, lane -> M = 16*fm + r + 8*(lane>>4), N = lane&15
    const float bv0 = convB[nTile0 + wn * 32 + lrow];
    const float bv1 = convB[nTile0 + wn * 32 + 16 + lrow];

#pragma unroll
    for (int fn = 0; fn < 2; ++fn) {
        float lm = -2.0f;
        const float bias = fn ? bv1 : bv0;
#pragma unroll
        for (int fm = 0; fm < 2; ++fm) {
            const v8f& acc = fm ? (fn ? c11 : c10) : (fn ? c01 : c00);
#pragma unroll
            for (int r = 0; r < 8; ++r) {
                const int t = wm * 32 + fm * 16 + r + 8 * lhi;
                if (t <= tmax) {
                    const float z = tanhf(acc[r] + bias);
                    lm = fmaxf(lm, z);
                }
            }
        }
        lm = fmaxf(lm, __shfl_xor(lm, 16, 32));   // combine lane-halves (same col)
        if (lhi == 0)
            red[wm * 64 + wn * 32 + fn * 16 + lrow] = lm;
    }
    __syncthreads();

    if (tid < 64) {
        float m = fmaxf(fmaxf(red[tid], red[64 + tid]),
                        fmaxf(red[128 + tid], red[192 + tid]));
        sf[(size_t)b * (3 * FN_) + sfOff + nTile0 + tid] = m;
    }
}

// -------------------------------------------------------------------------
// 4) Span means + boundary embeddings -> o[0:1800]
// -------------------------------------------------------------------------
__global__ void spans_kernel(const int* __restrict__ inputs,
                             const int* __restrict__ e1s, const int* __restrict__ e1e,
                             const int* __restrict__ e2s, const int* __restrict__ e2e,
                             const float* __restrict__ emb,
                             float* __restrict__ o) {
    const int b = blockIdx.x;
    const int e = threadIdx.x;
    if (e >= E_) return;
    const int s1 = e1s[b], t1 = e1e[b], s2 = e2s[b], t2 = e2e[b];
    float* ob = o + (size_t)b * OD_;

    float sum = 0.f;
    for (int j = s1; j <= t1; ++j) sum += emb[(size_t)inputs[b * L_ + j] * E_ + e];
    ob[e] = sum / (float)(t1 - s1 + 1);

    sum = 0.f;
    for (int j = s2; j <= t2; ++j) sum += emb[(size_t)inputs[b * L_ + j] * E_ + e];
    ob[E_ + e] = sum / (float)(t2 - s2 + 1);

    ob[2 * E_ + e] = emb[(size_t)inputs[b * L_ + (s1 - 1)] * E_ + e];
    ob[3 * E_ + e] = emb[(size_t)inputs[b * L_ + (t1 + 1)] * E_ + e];
    ob[4 * E_ + e] = emb[(size_t)inputs[b * L_ + (s2 - 1)] * E_ + e];
    ob[5 * E_ + e] = emb[(size_t)inputs[b * L_ + (t2 + 1)] * E_ + e];
}

// -------------------------------------------------------------------------
// 5) FC1: g = tanh(sf @ W1^T + b1)  -> o[1800:1900]
// -------------------------------------------------------------------------
__global__ void fc1_kernel(const float* __restrict__ sf,
                           const float* __restrict__ W1,
                           const float* __restrict__ b1,
                           float* __restrict__ o) {
    const int b = blockIdx.x;
    const int h = threadIdx.x;
    if (h >= H2_) return;
    float acc = b1[h];
    const float* s = sf + (size_t)b * (3 * FN_);
    const float* w = W1 + (size_t)h * (3 * FN_);
    for (int j = 0; j < 3 * FN_; ++j) acc += s[j] * w[j];
    o[(size_t)b * OD_ + 6 * E_ + h] = tanhf(acc);
}

// -------------------------------------------------------------------------
// 6) FC2: y = o @ W2^T + b2
// -------------------------------------------------------------------------
__global__ void fc2_kernel(const float* __restrict__ o,
                           const float* __restrict__ W2,
                           const float* __restrict__ b2,
                           float* __restrict__ y) {
    const int b   = blockIdx.x;
    const int lab = threadIdx.x;
    if (lab >= LAB_) return;
    float acc = b2[lab];
    const float* ob = o  + (size_t)b * OD_;
    const float* w  = W2 + (size_t)lab * OD_;
    for (int j = 0; j < OD_; ++j) acc += ob[j] * w[j];
    y[(size_t)b * LAB_ + lab] = acc;
}

// -------------------------------------------------------------------------
extern "C" void kernel_launch(void* const* d_in, const int* in_sizes, int n_in,
                              void* d_out, int out_size, void* d_ws, size_t ws_size,
                              hipStream_t stream) {
    const int*   inputs = (const int*)d_in[0];
    const int*   e1s    = (const int*)d_in[1];
    const int*   e1e    = (const int*)d_in[2];
    const int*   e2s    = (const int*)d_in[3];
    const int*   e2e    = (const int*)d_in[4];
    const int*   p1     = (const int*)d_in[5];
    const int*   p2     = (const int*)d_in[6];
    const float* emb    = (const float*)d_in[7];
    const float* pos1   = (const float*)d_in[8];
    const float* pos2   = (const float*)d_in[9];
    const float* cw3    = (const float*)d_in[10];
    const float* cb3    = (const float*)d_in[11];
    const float* cw4    = (const float*)d_in[12];
    const float* cb4    = (const float*)d_in[13];
    const float* cw5    = (const float*)d_in[14];
    const float* cb5    = (const float*)d_in[15];
    const float* W1     = (const float*)d_in[16];
    const float* b1     = (const float*)d_in[17];
    const float* W2     = (const float*)d_in[18];
    const float* b2     = (const float*)d_in[19];
    float* y = (float*)d_out;

    // workspace layout (all sizes multiples of 1024B)
    char* ws = (char*)d_ws;
    const size_t embedBytes = (size_t)(B_ * L_ + 8) * FDP_ * sizeof(_Float16);
    const size_t w3Bytes    = (size_t)FN_ * 3 * FDP_ * sizeof(_Float16);
    const size_t w4Bytes    = (size_t)FN_ * 4 * FDP_ * sizeof(_Float16);
    const size_t w5Bytes    = (size_t)FN_ * 5 * FDP_ * sizeof(_Float16);
    const size_t sfBytes    = (size_t)B_ * 3 * FN_ * sizeof(float);

    _Float16* embedH = (_Float16*)ws;                 ws += embedBytes;
    _Float16* w3H    = (_Float16*)ws;                 ws += w3Bytes;
    _Float16* w4H    = (_Float16*)ws;                 ws += w4Bytes;
    _Float16* w5H    = (_Float16*)ws;                 ws += w5Bytes;
    float*    sf     = (float*)ws;                    ws += sfBytes;
    float*    o      = (float*)ws;

    // 1) embed (f16) incl. zeroed pad rows
    build_embed_kernel<<<B_ * L_ + 8, 256, 0, stream>>>(inputs, p1, p2, emb,
                                                        pos1, pos2, embedH);
    // 2) weight repack to f16, K padded to k*1024
    repack_w_kernel<<<FN_ * 3, 256, 0, stream>>>(cw3, w3H, 3);
    repack_w_kernel<<<FN_ * 4, 256, 0, stream>>>(cw4, w4H, 4);
    repack_w_kernel<<<FN_ * 5, 256, 0, stream>>>(cw5, w5H, 5);

    // 3) WMMA implicit-GEMM convs + tanh + max -> sf
    dim3 gg(FN_ / 64, B_);
    conv_gemm_kernel<<<gg, 256, 0, stream>>>(embedH, w3H, cb3, sf, 3, 0);
    conv_gemm_kernel<<<gg, 256, 0, stream>>>(embedH, w4H, cb4, sf, 4, FN_);
    conv_gemm_kernel<<<gg, 256, 0, stream>>>(embedH, w5H, cb5, sf, 5, 2 * FN_);

    // 4) lexical features -> o[0:1800]
    spans_kernel<<<B_, 320, 0, stream>>>(inputs, e1s, e1e, e2s, e2e, emb, o);
    // 5) FC1 -> o[1800:1900]
    fc1_kernel<<<B_, 128, 0, stream>>>(sf, W1, b1, o);
    // 6) FC2 -> y
    fc2_kernel<<<B_, 32, 0, stream>>>(o, W2, b2, y);
    (void)in_sizes; (void)n_in; (void)out_size; (void)ws_size;
}
